// MockMoEModule_2937757631191
// MI455X (gfx1250) — matrix operand order repaired and checked by reading, not compiled, not run on hardware
//
#include <hip/hip_runtime.h>

// ---------------------------------------------------------------------------
// Grouped-GEMM SwiGLU expert FFN for MI455X (gfx1250, wave32, WMMA bf16)
//   per expert e:  y = (silu(x @ gate_e^T) * (x @ up_e^T)) @ down_e^T
// Fused kernel: x tile (97KB) + full h tile (193KB) live in LDS (320KB/WGP).
// Weights pre-converted f32->bf16 into d_ws once per launch (f32 fallback).
// Per-wave tile: 48 M x 32 N -> 12 WMMAs per k-step sharing 3 A-frags.
// ---------------------------------------------------------------------------

typedef __bf16 bf16;
typedef __attribute__((ext_vector_type(16))) bf16  bf16x16;
typedef __attribute__((ext_vector_type(8)))  bf16  bf16x8;
typedef __attribute__((ext_vector_type(4)))  bf16  bf16x4;
typedef __attribute__((ext_vector_type(8)))  float f32x8;
typedef __attribute__((ext_vector_type(4)))  float f32x4;

#define H_DIM    1024
#define I_DIM    2048
#define TILE_M   48            // 3 M-frags of 16
#define XS       (H_DIM + 8)   // padded LDS row stride (bf16 elems) for x tile
#define HS       (I_DIM + 8)   // padded LDS row stride (bf16 elems) for h tile
#define NTHREADS 256           // 8 wave32s

__device__ __forceinline__ f32x8 wmma_bf16(bf16x16 a, bf16x16 b, f32x8 c) {
  return __builtin_amdgcn_wmma_f32_16x16x32_bf16(
      false, a, false, b, (short)0, c, false, false);
}

// A-frag (16-bit A 16x32 layout): lane m = L%16, half = L/16.
// VGPRs 0-7 hold k = half*16..+15 (contiguous, matching B)
// -> two contiguous 16B runs in a row-major LDS tile -> 2x ds_load_b128.
__device__ __forceinline__ bf16x16 load_a_frag(const bf16* base, int stride,
                                               int m, int k, int half) {
  union { bf16x16 v; bf16x8 h[2]; } u;
  u.h[0] = *(const bf16x8*)(base + m * stride + k + half * 16);
  u.h[1] = *(const bf16x8*)(base + m * stride + k + half * 16 + 8);
  return u.v;
}

// B-frag (32x16): lane col n = L%16, half = L/16 selects k = half*16..+15,
// contiguous in memory (weights are K-contiguous because of the @W^T form).
__device__ __forceinline__ bf16x16 load_b_bf16(const bf16* __restrict__ p) {
  union { bf16x16 v; bf16x8 h[2]; } u;
  u.h[0] = ((const bf16x8*)p)[0];
  u.h[1] = ((const bf16x8*)p)[1];
  return u.v;
}

__device__ __forceinline__ bf16x16 load_b_f32(const float* __restrict__ p) {
  f32x4 f0 = ((const f32x4*)p)[0];
  f32x4 f1 = ((const f32x4*)p)[1];
  f32x4 f2 = ((const f32x4*)p)[2];
  f32x4 f3 = ((const f32x4*)p)[3];
  bf16x16 b;
#pragma unroll
  for (int i = 0; i < 4; ++i) {
    b[i]      = (bf16)f0[i];
    b[4 + i]  = (bf16)f1[i];
    b[8 + i]  = (bf16)f2[i];
    b[12 + i] = (bf16)f3[i];
  }
  return b;
}

// silu(g) = g * sigmoid(g); result is quantized to bf16, so a 1-ulp
// v_rcp_f32 is plenty (avoids the IEEE div_scale/div_fixup expansion).
__device__ __forceinline__ float silu_f(float g) {
  return g * __builtin_amdgcn_rcpf(1.0f + __expf(-g));
}

// ---- weight f32 -> bf16 conversion (runs once per launch) -----------------
__global__ void cvt_f32_to_bf16_kernel(const float* __restrict__ src,
                                       bf16* __restrict__ dst, long long n4) {
  long long i      = (long long)blockIdx.x * blockDim.x + threadIdx.x;
  long long stride = (long long)gridDim.x * blockDim.x;
  for (; i < n4; i += stride) {
    f32x4 v = ((const f32x4*)src)[i];
    bf16x4 b;
    b[0] = (bf16)v[0]; b[1] = (bf16)v[1];
    b[2] = (bf16)v[2]; b[3] = (bf16)v[3];
    ((bf16x4*)dst)[i] = b;
  }
}

// ---- fused MoE FFN --------------------------------------------------------
template <bool WB16>
__global__ __launch_bounds__(NTHREADS)
void moe_swiglu_ffn_kernel(const float* __restrict__ tokens,
                           const float* __restrict__ gate_f,
                           const float* __restrict__ up_f,
                           const float* __restrict__ down_f,
                           const bf16*  __restrict__ gate_b,
                           const bf16*  __restrict__ up_b,
                           const bf16*  __restrict__ down_b,
                           const void*  __restrict__ m_sizes_raw,
                           const void*  __restrict__ m_offsets_raw,
                           float* __restrict__ out) {
  extern __shared__ char smem[];
  bf16* xs = (bf16*)smem;                                   // [TILE_M][XS]
  bf16* hs = (bf16*)(smem + (size_t)TILE_M * XS * 2);       // [TILE_M][HS]

  const int e = blockIdx.y;

  // m_sizes/m_offsets: int64 per reference, int32 if JAX demoted them.
  const int* s32 = (const int*)m_sizes_raw;
  const bool is64 = (s32[1] == 0);
  long long off, sz;
  if (is64) {
    off = ((const long long*)m_offsets_raw)[e];
    sz  = ((const long long*)m_sizes_raw)[e];
  } else {
    off = (long long)((const int*)m_offsets_raw)[e];
    sz  = (long long)((const int*)m_sizes_raw)[e];
  }

  const long long tile0 = (long long)blockIdx.x * TILE_M;
  if (tile0 >= sz) return;                         // uniform early exit
  const long long rem = sz - tile0;
  const int rows_valid = (int)(rem < TILE_M ? rem : (long long)TILE_M);
  const size_t row0 = (size_t)(off + tile0);

  const int tid  = threadIdx.x;
  const int wave = tid >> 5;
  const int lane = tid & 31;
  const int half = lane >> 4;
  const int l16  = lane & 15;

  // ---- stage x tile (f32 -> bf16) into LDS --------------------------------
  for (int idx = tid; idx < TILE_M * (H_DIM / 4); idx += NTHREADS) {
    const int r  = idx >> 8;          // H_DIM/4 == 256 float4 per row
    const int c4 = idx & 255;
    f32x4 v = {0.f, 0.f, 0.f, 0.f};
    if (r < rows_valid)
      v = *(const f32x4*)(tokens + (row0 + (size_t)r) * H_DIM + c4 * 4);
    bf16x4 b;
    b[0] = (bf16)v[0]; b[1] = (bf16)v[1];
    b[2] = (bf16)v[2]; b[3] = (bf16)v[3];
    *(bf16x4*)(xs + r * XS + c4 * 4) = b;
  }
  __syncthreads();

  // ---- phase 1: h = silu(x @ gate^T) * (x @ up^T) -> bf16 in LDS ----------
  // Each wave: 48M x 32N (two 16-col chunks) per outer iteration.
  const size_t wbase = (size_t)e * I_DIM * H_DIM;
#pragma unroll 1
  for (int nc = 0; nc < (I_DIM / 8) / 32; ++nc) {  // 8 iterations of 32 cols
    const int n0 = wave * (I_DIM / 8) + nc * 32;
    const size_t wrow = wbase + (size_t)(n0 + l16) * H_DIM + half * 16;
    f32x8 accg[2][3], accu[2][3];
#pragma unroll
    for (int c = 0; c < 2; ++c)
#pragma unroll
      for (int r = 0; r < 3; ++r) { accg[c][r] = 0.f; accu[c][r] = 0.f; }

#pragma unroll 1
    for (int k0 = 0; k0 < H_DIM; k0 += 32) {
      bf16x16 a[3];
      a[0] = load_a_frag(xs, XS, l16,      k0, half);
      a[1] = load_a_frag(xs, XS, l16 + 16, k0, half);
      a[2] = load_a_frag(xs, XS, l16 + 32, k0, half);
#pragma unroll
      for (int c = 0; c < 2; ++c) {
        const size_t wr = wrow + (size_t)(c * 16) * H_DIM + k0;
        bf16x16 bg, bu;
        if (WB16) {
          bg = load_b_bf16(gate_b + wr);
          bu = load_b_bf16(up_b   + wr);
        } else {
          bg = load_b_f32(gate_f + wr);
          bu = load_b_f32(up_f   + wr);
        }
#pragma unroll
        for (int r = 0; r < 3; ++r) {
          accg[c][r] = wmma_bf16(a[r], bg, accg[c][r]);
          accu[c][r] = wmma_bf16(a[r], bu, accu[c][r]);
        }
      }
    }
    // epilogue: SiLU(g)*u -> bf16 h tile.  C/D layout: M = v + 8*half + 16*r
#pragma unroll
    for (int c = 0; c < 2; ++c) {
      const int col = n0 + c * 16 + l16;
#pragma unroll
      for (int r = 0; r < 3; ++r) {
#pragma unroll
        for (int v = 0; v < 8; ++v) {
          const int m = v + 8 * half + 16 * r;
          hs[m * HS + col] = (bf16)(silu_f(accg[c][r][v]) * accu[c][r][v]);
        }
      }
    }
  }
  __syncthreads();

  // ---- phase 2: out = h @ down^T ------------------------------------------
  const size_t dbase = (size_t)e * H_DIM * I_DIM;
#pragma unroll 1
  for (int nc = 0; nc < (H_DIM / 8) / 32; ++nc) {  // 4 iterations of 32 cols
    const int n0 = wave * (H_DIM / 8) + nc * 32;
    const size_t drow = dbase + (size_t)(n0 + l16) * I_DIM + half * 16;
    f32x8 acc[2][3];
#pragma unroll
    for (int c = 0; c < 2; ++c)
#pragma unroll
      for (int r = 0; r < 3; ++r) acc[c][r] = 0.f;

#pragma unroll 2
    for (int k0 = 0; k0 < I_DIM; k0 += 32) {
      bf16x16 a[3];
      a[0] = load_a_frag(hs, HS, l16,      k0, half);
      a[1] = load_a_frag(hs, HS, l16 + 16, k0, half);
      a[2] = load_a_frag(hs, HS, l16 + 32, k0, half);
#pragma unroll
      for (int c = 0; c < 2; ++c) {
        const size_t dr = drow + (size_t)(c * 16) * I_DIM + k0;
        bf16x16 bd;
        if (WB16) bd = load_b_bf16(down_b + dr);
        else      bd = load_b_f32(down_f + dr);
#pragma unroll
        for (int r = 0; r < 3; ++r)
          acc[c][r] = wmma_bf16(a[r], bd, acc[c][r]);
      }
    }
#pragma unroll
    for (int c = 0; c < 2; ++c) {
      const int nglob = n0 + c * 16 + l16;
#pragma unroll
      for (int r = 0; r < 3; ++r) {
#pragma unroll
        for (int v = 0; v < 8; ++v) {
          const int m = v + 8 * half + 16 * r;
          if (m < rows_valid)
            out[(row0 + (size_t)m) * H_DIM + nglob] = acc[c][r][v];
        }
      }
    }
  }
}

extern "C" void kernel_launch(void* const* d_in, const int* in_sizes, int n_in,
                              void* d_out, int out_size, void* d_ws, size_t ws_size,
                              hipStream_t stream) {
  const float* tokens = (const float*)d_in[0];
  const float* gate_f = (const float*)d_in[1];
  const float* up_f   = (const float*)d_in[2];
  const float* down_f = (const float*)d_in[3];
  const void*  m_sz   = d_in[4];
  const void*  m_off  = d_in[5];
  float* out = (float*)d_out;

  const int T = in_sizes[0] / H_DIM;   // 65536
  const int E = in_sizes[4];           // 16
  (void)n_in; (void)out_size;

  dim3 grid((unsigned)((T + TILE_M - 1) / TILE_M), (unsigned)E);
  const size_t smem = (size_t)TILE_M * XS * 2 + (size_t)TILE_M * HS * 2; // ~290 KB

  const size_t welem = (size_t)E * I_DIM * H_DIM;     // elems per weight tensor
  const size_t need  = 3 * welem * 2;                  // bf16 bytes for 3 tensors

  if (ws_size >= need) {
    bf16* gate_b = (bf16*)d_ws;
    bf16* up_b   = gate_b + welem;
    bf16* down_b = up_b + welem;
    const long long n4 = (long long)(welem / 4);
    cvt_f32_to_bf16_kernel<<<8192, 256, 0, stream>>>(gate_f, gate_b, n4);
    cvt_f32_to_bf16_kernel<<<8192, 256, 0, stream>>>(up_f,   up_b,   n4);
    cvt_f32_to_bf16_kernel<<<8192, 256, 0, stream>>>(down_f, down_b, n4);
    moe_swiglu_ffn_kernel<true><<<grid, NTHREADS, smem, stream>>>(
        tokens, nullptr, nullptr, nullptr, gate_b, up_b, down_b,
        m_sz, m_off, out);
  } else {
    moe_swiglu_ffn_kernel<false><<<grid, NTHREADS, smem, stream>>>(
        tokens, gate_f, up_f, down_f, nullptr, nullptr, nullptr,
        m_sz, m_off, out);
  }
}
